// GAT_82746839924854
// MI455X (gfx1250) — matrix operand order repaired and checked by reading, not compiled, not run on hardware
//
#include <hip/hip_runtime.h>
#include <hip/hip_bf16.h>

typedef __attribute__((ext_vector_type(16))) _Float16 v16h;
typedef __attribute__((ext_vector_type(8)))  _Float16 v8h;
typedef __attribute__((ext_vector_type(8)))  float    v8f;

constexpr int NN = 100000;   // nodes
constexpr int EE = 800000;   // raw edges
constexpr int EP = 900000;   // edges + self loops
constexpr int GG = 1024;     // graphs
constexpr int F1 = 384;      // 6*64 input feature
constexpr int HC = 256;      // heads*channels for both layers (2*128, 4*64)

__device__ __forceinline__ float sigm(float x) { return 1.0f / (1.0f + __expf(-x)); }
__device__ __forceinline__ float lrelu(float x) { return x > 0.0f ? x : 0.2f * x; }

__device__ __forceinline__ void edge_ends(const int* __restrict__ ei, int e, int& s, int& d) {
    if (e < EE) { s = ei[e]; d = ei[EE + e]; }
    else        { s = e - EE; d = s; }
}

// float atomic max via sign-aware integer atomics
__device__ __forceinline__ void atomicMaxFloat(float* addr, float val) {
    if (val >= 0.0f) atomicMax((int*)addr, __float_as_int(val));
    else             atomicMin((unsigned int*)addr, __float_as_uint(val));
}

// ---------------- fill ----------------
__global__ void fill_f32(float* __restrict__ p, int n, float v) {
    int t = blockIdx.x * blockDim.x + threadIdx.x;
    if (t < n) p[t] = v;
}

// ---------------- weight transpose + f16 convert: W[K,NC] f32 -> WT[NC,K] f16 ----------------
__global__ void wt_f16(const float* __restrict__ W, _Float16* __restrict__ WT, int K, int NC) {
    int t = blockIdx.x * blockDim.x + threadIdx.x;
    if (t >= K * NC) return;
    int n = t / K, k = t % K;
    WT[t] = (_Float16)W[k * NC + n];
}

// ---------------- embedding gather -> f16 activation matrix [NN, 384] ----------------
__global__ void embed_f16(const int* __restrict__ x,
                          const float* __restrict__ e0, const float* __restrict__ e1,
                          const float* __restrict__ e2, const float* __restrict__ e3,
                          const float* __restrict__ e4, const float* __restrict__ e5,
                          _Float16* __restrict__ out) {
    int t = blockIdx.x * blockDim.x + threadIdx.x;
    if (t >= NN * F1) return;
    int n = t / F1, r = t % F1;
    int i = r >> 6, c = r & 63;
    int idx = x[n * 6 + i];
    const float* tbl;
    switch (i) {
        case 0: tbl = e0; break; case 1: tbl = e1; break; case 2: tbl = e2; break;
        case 3: tbl = e3; break; case 4: tbl = e4; break; default: tbl = e5; break;
    }
    out[t] = (_Float16)tbl[idx * 64 + c];
}

// ---------------- WMMA GEMM: A[M,K] f16 x BT[NC,K] f16 -> C[M,NC] f32 ----------------
// One wave computes a 16x64 output strip (4 accumulators): the A fragment is
// loaded once per K-step and reused against 4 B fragments -> 4x less A traffic.
__global__ void gemm_wmma(const _Float16* __restrict__ A,
                          const _Float16* __restrict__ BT,
                          float* __restrict__ Cout,
                          int K, int NC, int numTilesM) {
    int wave = blockIdx.x * (blockDim.x >> 5) + (threadIdx.x >> 5);
    int stripsPerRow = NC >> 6;                  // 64-wide N strips
    int tileM = wave / stripsPerRow;
    int strip = wave % stripsPerRow;
    if (tileM >= numTilesM) return;

    int lane = threadIdx.x & 31;
    int m  = (tileM << 4) + (lane & 15);         // A row for this lane
    int nb = (strip << 6) + (lane & 15);         // first B column / C column for this lane
    int k0 = (lane >> 4) << 4;                   // A: lanes 16-31 hold K+16..31
    int kb = (lane >> 4) << 4;                   // B: lanes 16-31 hold K+16..31

    const _Float16* arow = A + (size_t)m * K;
    const _Float16* b0r  = BT + (size_t)(nb     ) * K;
    const _Float16* b1r  = BT + (size_t)(nb + 16) * K;
    const _Float16* b2r  = BT + (size_t)(nb + 32) * K;
    const _Float16* b3r  = BT + (size_t)(nb + 48) * K;

    v8f c0 = {}, c1 = {}, c2 = {}, c3 = {};
    for (int k = 0; k < K; k += 32) {
        v8h alo = *(const v8h*)(arow + k + k0);
        v8h ahi = *(const v8h*)(arow + k + k0 + 8);
        v16h a = __builtin_shufflevector(alo, ahi, 0,1,2,3,4,5,6,7,8,9,10,11,12,13,14,15);

        v8h l0 = *(const v8h*)(b0r + k + kb), h0 = *(const v8h*)(b0r + k + kb + 8);
        v8h l1 = *(const v8h*)(b1r + k + kb), h1 = *(const v8h*)(b1r + k + kb + 8);
        v8h l2 = *(const v8h*)(b2r + k + kb), h2 = *(const v8h*)(b2r + k + kb + 8);
        v8h l3 = *(const v8h*)(b3r + k + kb), h3 = *(const v8h*)(b3r + k + kb + 8);
        v16h bf0 = __builtin_shufflevector(l0, h0, 0,1,2,3,4,5,6,7,8,9,10,11,12,13,14,15);
        v16h bf1 = __builtin_shufflevector(l1, h1, 0,1,2,3,4,5,6,7,8,9,10,11,12,13,14,15);
        v16h bf2 = __builtin_shufflevector(l2, h2, 0,1,2,3,4,5,6,7,8,9,10,11,12,13,14,15);
        v16h bf3 = __builtin_shufflevector(l3, h3, 0,1,2,3,4,5,6,7,8,9,10,11,12,13,14,15);

        c0 = __builtin_amdgcn_wmma_f32_16x16x32_f16(false, a, false, bf0, (short)0, c0, false, false);
        c1 = __builtin_amdgcn_wmma_f32_16x16x32_f16(false, a, false, bf1, (short)0, c1, false, false);
        c2 = __builtin_amdgcn_wmma_f32_16x16x32_f16(false, a, false, bf2, (short)0, c2, false, false);
        c3 = __builtin_amdgcn_wmma_f32_16x16x32_f16(false, a, false, bf3, (short)0, c3, false, false);
    }

    int rbase = (tileM << 4) + ((lane >> 4) << 3);   // lanes 16-31 hold rows M=8..15
#pragma unroll
    for (int r = 0; r < 8; ++r) {
        float* crow = Cout + (size_t)(rbase + r) * NC + nb;
        crow[0]  = c0[r];
        crow[16] = c1[r];
        crow[32] = c2[r];
        crow[48] = c3[r];
    }
}

// ---------------- per-node attention logits: s = <xw, a_src>, d = <xw, a_dst> ----------------
__global__ void attn_node(const float* __restrict__ xw,
                          const float* __restrict__ a_src, const float* __restrict__ a_dst,
                          float* __restrict__ S, float* __restrict__ D, int H, int C) {
    int t = blockIdx.x * blockDim.x + threadIdx.x;
    if (t >= NN * H) return;
    int n = t / H, h = t % H;
    const float* xr = xw + (size_t)n * HC + h * C;
    const float* as = a_src + h * C;
    const float* ad = a_dst + h * C;
    float s = 0.0f, d = 0.0f;
#pragma unroll 4
    for (int c = 0; c < C; ++c) { float v = xr[c]; s += v * as[c]; d += v * ad[c]; }
    S[t] = s; D[t] = d;
}

// ---------------- edge pass A: segment max of leaky-relu logits over dst ----------------
__global__ void edge_max(const int* __restrict__ ei, const float* __restrict__ S,
                         const float* __restrict__ D, float* __restrict__ emax, int H) {
    int t = blockIdx.x * blockDim.x + threadIdx.x;
    if (t >= EP * H) return;
    int e = t / H, h = t % H;
    int s, d; edge_ends(ei, e, s, d);
    float v = lrelu(S[s * H + h] + D[d * H + h]);
    atomicMaxFloat(&emax[d * H + h], v);
}

// ---------------- edge pass B: softmax denominator ----------------
__global__ void edge_denom(const int* __restrict__ ei, const float* __restrict__ S,
                           const float* __restrict__ D, const float* __restrict__ emax,
                           float* __restrict__ denom, int H) {
    int t = blockIdx.x * blockDim.x + threadIdx.x;
    if (t >= EP * H) return;
    int e = t / H, h = t % H;
    int s, d; edge_ends(ei, e, s, d);
    int i = d * H + h;
    float v = lrelu(S[s * H + h] + D[i]);
    atomicAdd(&denom[i], __expf(v - emax[i]));
}

// ---------------- edge pass C: alpha-weighted scatter-add of xw[src] ----------------
__global__ void edge_agg(const int* __restrict__ ei, const float* __restrict__ S,
                         const float* __restrict__ D, const float* __restrict__ emax,
                         const float* __restrict__ denom, const float* __restrict__ xw,
                         float* __restrict__ agg, int H, int C) {
    int t = blockIdx.x * blockDim.x + threadIdx.x;     // EP * 64 threads, 4 chans each
    if (t >= EP * 64) return;
    int e = t >> 6;
    int ch = (t & 63) << 2;
    int s, d; edge_ends(ei, e, s, d);
    int h = ch / C;
    int i = d * H + h;
    float v = lrelu(S[s * H + h] + D[i]);
    float alpha = __expf(v - emax[i]) / (denom[i] + 1e-16f);
    const float4 xv = *(const float4*)(xw + (size_t)s * HC + ch);
    float* ap = agg + (size_t)d * HC + ch;
    atomicAdd(ap + 0, alpha * xv.x);
    atomicAdd(ap + 1, alpha * xv.y);
    atomicAdd(ap + 2, alpha * xv.z);
    atomicAdd(ap + 3, alpha * xv.w);
}

// ---------------- layer-1 epilogue: head-mean + bias + sigmoid -> f16 [NN,128] ----------------
__global__ void finalize_h1(const float* __restrict__ agg, const float* __restrict__ b1,
                            _Float16* __restrict__ out) {
    int t = blockIdx.x * blockDim.x + threadIdx.x;
    if (t >= NN * 128) return;
    int n = t >> 7, c = t & 127;
    float v = 0.5f * (agg[(size_t)n * HC + c] + agg[(size_t)n * HC + 128 + c]) + b1[c];
    out[t] = (_Float16)sigm(v);
}

// ---------------- layer-2 epilogue fused with global_add_pool ----------------
__global__ void finalize_pool(const float* __restrict__ agg, const float* __restrict__ b2,
                              const int* __restrict__ batch, float* __restrict__ gpool) {
    int t = blockIdx.x * blockDim.x + threadIdx.x;
    if (t >= NN * 64) return;
    int n = t >> 6, c = t & 63;
    const float* a = agg + (size_t)n * HC;
    float v = 0.25f * (a[c] + a[64 + c] + a[128 + c] + a[192 + c]) + b2[c];
    atomicAdd(&gpool[batch[n] * 64 + c], sigm(v));
}

// ---------------- graph head: [G,64] @ [64,1] + b, sigmoid ----------------
__global__ void graph_head(const float* __restrict__ gpool, const float* __restrict__ lw,
                           const float* __restrict__ lb, float* __restrict__ out) {
    int g = blockIdx.x * blockDim.x + threadIdx.x;
    if (g >= GG) return;
    float acc = lb[0];
    const float* gr = gpool + g * 64;
#pragma unroll 4
    for (int c = 0; c < 64; ++c) acc += gr[c] * lw[c];
    out[g] = sigm(acc);
}

static inline int blks(long long n) { return (int)((n + 255) / 256); }

extern "C" void kernel_launch(void* const* d_in, const int* in_sizes, int n_in,
                              void* d_out, int out_size, void* d_ws, size_t ws_size,
                              hipStream_t stream) {
    const int*   x      = (const int*)  d_in[0];
    const int*   eidx   = (const int*)  d_in[1];
    const int*   batch  = (const int*)  d_in[3];
    const float* emb[6] = { (const float*)d_in[4], (const float*)d_in[5], (const float*)d_in[6],
                            (const float*)d_in[7], (const float*)d_in[8], (const float*)d_in[9] };
    const float* W1     = (const float*)d_in[10];
    const float* as1    = (const float*)d_in[11];
    const float* ad1    = (const float*)d_in[12];
    const float* b1     = (const float*)d_in[13];
    const float* W2     = (const float*)d_in[14];
    const float* as2    = (const float*)d_in[15];
    const float* ad2    = (const float*)d_in[16];
    const float* b2     = (const float*)d_in[17];
    const float* lw     = (const float*)d_in[18];
    const float* lb     = (const float*)d_in[19];
    float* out          = (float*)d_out;

    // workspace carve-up (256B aligned slabs)
    char* ws = (char*)d_ws;
    size_t off = 0;
    auto take = [&](size_t bytes) -> void* {
        void* p = ws + off;
        off = (off + bytes + 255) & ~(size_t)255;
        return p;
    };
    _Float16* A16   = (_Float16*)take((size_t)NN * F1 * 2);   // x_f16 [N,384]; reused as h1_f16 [N,128]
    _Float16* W1T   = (_Float16*)take((size_t)256 * F1 * 2);  // W1^T f16 [256,384]
    _Float16* W2T   = (_Float16*)take((size_t)256 * 128 * 2); // W2^T f16 [256,128]
    float*    XW    = (float*)take((size_t)NN * HC * 4);      // xw [N,256] (both layers)
    float*    AGG   = (float*)take((size_t)NN * HC * 4);      // segment-sum accumulator
    float*    Sb    = (float*)take((size_t)NN * 4 * 4);
    float*    Db    = (float*)take((size_t)NN * 4 * 4);
    float*    EMAXb = (float*)take((size_t)NN * 4 * 4);
    float*    DENb  = (float*)take((size_t)NN * 4 * 4);
    float*    GP    = (float*)take((size_t)GG * 64 * 4);

    const float NEGINF = -__builtin_inff();

    // ---- prep: weights to f16 transposed, embeddings to f16 ----
    wt_f16<<<blks((long long)F1 * 256), 256, 0, stream>>>(W1, W1T, F1, 256);
    wt_f16<<<blks((long long)128 * 256), 256, 0, stream>>>(W2, W2T, 128, 256);
    embed_f16<<<blks((long long)NN * F1), 256, 0, stream>>>(x, emb[0], emb[1], emb[2],
                                                            emb[3], emb[4], emb[5], A16);

    // GEMM grid: (NN/16) M-tiles x (256/64) N-strips, 8 waves per block
    const int gemmBlocks = (NN / 16) * 4 / 8;

    // ================= layer 1: H=2, C=128 =================
    gemm_wmma<<<gemmBlocks, 256, 0, stream>>>(A16, W1T, XW, F1, 256, NN / 16);
    attn_node<<<blks((long long)NN * 2), 256, 0, stream>>>(XW, as1, ad1, Sb, Db, 2, 128);
    fill_f32<<<blks((long long)NN * 2), 256, 0, stream>>>(EMAXb, NN * 2, NEGINF);
    fill_f32<<<blks((long long)NN * 2), 256, 0, stream>>>(DENb, NN * 2, 0.0f);
    fill_f32<<<blks((long long)NN * HC), 256, 0, stream>>>(AGG, NN * HC, 0.0f);
    edge_max  <<<blks((long long)EP * 2), 256, 0, stream>>>(eidx, Sb, Db, EMAXb, 2);
    edge_denom<<<blks((long long)EP * 2), 256, 0, stream>>>(eidx, Sb, Db, EMAXb, DENb, 2);
    edge_agg  <<<blks((long long)EP * 64), 256, 0, stream>>>(eidx, Sb, Db, EMAXb, DENb, XW, AGG, 2, 128);
    finalize_h1<<<blks((long long)NN * 128), 256, 0, stream>>>(AGG, b1, A16);

    // ================= layer 2: H=4, C=64 =================
    gemm_wmma<<<gemmBlocks, 256, 0, stream>>>(A16, W2T, XW, 128, 256, NN / 16);
    attn_node<<<blks((long long)NN * 4), 256, 0, stream>>>(XW, as2, ad2, Sb, Db, 4, 64);
    fill_f32<<<blks((long long)NN * 4), 256, 0, stream>>>(EMAXb, NN * 4, NEGINF);
    fill_f32<<<blks((long long)NN * 4), 256, 0, stream>>>(DENb, NN * 4, 0.0f);
    fill_f32<<<blks((long long)NN * HC), 256, 0, stream>>>(AGG, NN * HC, 0.0f);
    fill_f32<<<blks((long long)GG * 64), 256, 0, stream>>>(GP, GG * 64, 0.0f);
    edge_max  <<<blks((long long)EP * 4), 256, 0, stream>>>(eidx, Sb, Db, EMAXb, 4);
    edge_denom<<<blks((long long)EP * 4), 256, 0, stream>>>(eidx, Sb, Db, EMAXb, DENb, 4);
    edge_agg  <<<blks((long long)EP * 64), 256, 0, stream>>>(eidx, Sb, Db, EMAXb, DENb, XW, AGG, 4, 64);
    finalize_pool<<<blks((long long)NN * 64), 256, 0, stream>>>(AGG, b2, batch, GP);

    // ================= graph head =================
    graph_head<<<blks(GG), 256, 0, stream>>>(GP, lw, lb, out);
}
